// IntentSlotLabellingModel_5815385719211
// MI455X (gfx1250) — matrix operand order repaired and checked by reading, not compiled, not run on hardware
//
#include <hip/hip_runtime.h>
#include <hip/hip_bf16.h>
#include <stdint.h>

typedef __attribute__((ext_vector_type(16))) __bf16 v16bf;
typedef __attribute__((ext_vector_type(8)))  float  v8f;
typedef unsigned short u16;
typedef unsigned int   u32;

__device__ __forceinline__ u16 f2bf(float f) {
  union { float f; u32 u; } c; c.f = f;
  u32 x = c.u;
  x += 0x7FFFu + ((x >> 16) & 1u);   // round-to-nearest-even
  return (u16)(x >> 16);
}

// ---------------- gather embedding rows, convert f32 -> bf16 ----------------
__global__ __launch_bounds__(256) void gather_embed_bf16(
    const int* __restrict__ tok, const float* __restrict__ emb,
    u16* __restrict__ X)
{
  int e   = blockIdx.x * 256 + threadIdx.x;  // chunk id (8 elems)
  int row = e >> 6;                          // 512/8 = 64 chunks per row
  int col = (e & 63) << 3;
  int t   = tok[row];
  const float* src = emb + (size_t)t * 512 + col;
  float4 f0 = ((const float4*)src)[0];
  float4 f1 = ((const float4*)src)[1];
  union { u16 h[8]; uint4 q; } o;
  o.h[0] = f2bf(f0.x); o.h[1] = f2bf(f0.y); o.h[2] = f2bf(f0.z); o.h[3] = f2bf(f0.w);
  o.h[4] = f2bf(f1.x); o.h[5] = f2bf(f1.y); o.h[6] = f2bf(f1.z); o.h[7] = f2bf(f1.w);
  *(uint4*)(X + (size_t)e * 8) = o.q;
}

// ------- pack-convert weights: in f32 [K,N] -> out u32 [K/2][N], pair (k, k+1) -------
// out[p*N + n] = bf16(in[2p][n]) | bf16(in[2p+1][n]) << 16   (WMMA B-operand pair layout)
__global__ __launch_bounds__(256) void pack_cvt_bf16(
    const float* __restrict__ in, u32* __restrict__ out, int P, int N)
{
  int i = blockIdx.x * 256 + threadIdx.x;
  if (i < P * N) {
    int p = i / N, n = i % N;
    float lo = in[(size_t)(2 * p) * N + n];
    float hi = in[(size_t)(2 * p + 1) * N + n];
    out[i] = (u32)f2bf(lo) | ((u32)f2bf(hi) << 16);
  }
}

// ------- pack-convert with transpose: in f32 [N,K] -> out u32 [K/2][N] -------
// out[p*N + n] = bf16(in[n][2p]) | bf16(in[n][2p+1]) << 16
__global__ __launch_bounds__(256) void pack_tcvt_bf16(
    const float* __restrict__ in, u32* __restrict__ out, int P, int N, int K)
{
  int i = blockIdx.x * 256 + threadIdx.x;
  if (i < P * N) {
    int p = i / N, n = i % N;
    float2 f = *(const float2*)(in + (size_t)n * K + 2 * p);
    out[i] = (u32)f2bf(f.x) | ((u32)f2bf(f.y) << 16);
  }
}

// ---------------- bf16 WMMA GEMM: Out = relu?(A[M,K] * Bp + bias) ----------------
// A: bf16 [M,K] row-major.  Bp: u32 [K/2][N] pre-packed k-pairs.
// Block tile 128x128, 8 waves (2x4), wave tile 64x32 (4x2 WMMA frags), K-step 32.
// Both tiles filled with GLOBAL_LOAD_ASYNC_TO_LDS_B128 (ASYNCcnt), double-buffered.
#define AS_STRIDE 48   // bf16 elements per A row in LDS (pad 32->48; keeps 16B alignment)

template<bool RELU, bool OUT_BF16>
__global__ __launch_bounds__(256) void gemm_bf16_wmma(
    const u16* __restrict__ A, const u32* __restrict__ Bp,
    const float* __restrict__ bias, void* __restrict__ OutP,
    int M, int N, int K)
{
  __shared__ u16 As[2][128 * AS_STRIDE];   // 24.0 KB
  __shared__ u32 Bs[2][16 * 128];          // 16.0 KB  (packed bf16 k-pairs)

  const int tid  = threadIdx.x;
  const int lane = tid & 31;
  const int wave = tid >> 5;
  const int wm   = wave >> 2;        // 0..1
  const int wn   = wave & 3;         // 0..3
  const int g    = lane >> 4;        // lane group (K half select)
  const int ln   = lane & 15;        // row/col within fragment

  const int bm = blockIdx.x * 128;
  const int bn = blockIdx.y * 128;

  // ---- A-tile async fill: thread t -> row ar, 16-elem half ah (32B/thread) ----
  const int ar = tid >> 1;
  const int ah = tid & 1;
  const int asIdx = ar * AS_STRIDE + ah * 16;
  const u32 ldsA0 = (u32)(uintptr_t)(&As[0][asIdx]);
  const u32 ldsA1 = (u32)(uintptr_t)(&As[1][asIdx]);
  const u32 gOffA = (u32)(((size_t)(bm + ar) * K + ah * 16) * 2);  // bytes

  // ---- B-tile async fill: thread t -> pair-row kp, 8 u32 columns at nb (32B/thread) ----
  const int kp = tid >> 4;           // 0..15
  const int nb = (tid & 15) * 8;
  const int bsIdx = kp * 128 + nb;
  const u32 ldsB0 = (u32)(uintptr_t)(&Bs[0][bsIdx]);
  const u32 ldsB1 = (u32)(uintptr_t)(&Bs[1][bsIdx]);
  const u32 gOffB = (u32)(((size_t)kp * N + bn + nb) * 4);         // bytes
  const u32 bTile = (u32)N * 64u;    // bytes per K-tile of Bp (16 rows * N * 4)

  const int nk = K >> 5;

  // ---- prologue: tile 0 into buffer 0 ----
  asm volatile("global_load_async_to_lds_b128 %0, %1, %2"
               :: "v"(ldsA0), "v"(gOffA), "s"(A) : "memory");
  asm volatile("global_load_async_to_lds_b128 %0, %1, %2"
               :: "v"(ldsA0 + 16u), "v"(gOffA + 16u), "s"(A) : "memory");
  asm volatile("global_load_async_to_lds_b128 %0, %1, %2"
               :: "v"(ldsB0), "v"(gOffB), "s"(Bp) : "memory");
  asm volatile("global_load_async_to_lds_b128 %0, %1, %2"
               :: "v"(ldsB0 + 16u), "v"(gOffB + 16u), "s"(Bp) : "memory");
  asm volatile("s_wait_asynccnt 0x0" ::: "memory");
  __syncthreads();

  v8f acc[4][2] = {};

  for (int kt = 0; kt < nk; ++kt) {
    const int  cur  = kt & 1;
    const bool more = (kt + 1) < nk;

    if (more) {                                     // async-prefetch next tile
      const u32 goA = gOffA + (u32)(kt + 1) * 64u;  // K-step = 32 elems = 64 bytes
      const u32 goB = gOffB + (u32)(kt + 1) * bTile;
      const u32 ldA = cur ? ldsA0 : ldsA1;          // next buffer
      const u32 ldB = cur ? ldsB0 : ldsB1;
      asm volatile("global_load_async_to_lds_b128 %0, %1, %2"
                   :: "v"(ldA), "v"(goA), "s"(A) : "memory");
      asm volatile("global_load_async_to_lds_b128 %0, %1, %2"
                   :: "v"(ldA + 16u), "v"(goA + 16u), "s"(A) : "memory");
      asm volatile("global_load_async_to_lds_b128 %0, %1, %2"
                   :: "v"(ldB), "v"(goB), "s"(Bp) : "memory");
      asm volatile("global_load_async_to_lds_b128 %0, %1, %2"
                   :: "v"(ldB + 16u), "v"(goB + 16u), "s"(Bp) : "memory");
    }

    // ---- fragments from LDS (contiguous k order: element s of lane half g holds K = g*16+s) ----
    v16bf afr[4], bfr[2];
    #pragma unroll
    for (int mt = 0; mt < 4; ++mt) {
      union { uint4 q[2]; v16bf v; } af;
      const u16* ap = &As[cur][(wm * 64 + mt * 16 + ln) * AS_STRIDE + g * 16];
      af.q[0] = ((const uint4*)ap)[0];          // K = g*16 .. g*16+7
      af.q[1] = ((const uint4*)(ap + 8))[0];    // K = g*16+8 .. g*16+15
      afr[mt] = af.v;
    }
    #pragma unroll
    for (int nt = 0; nt < 2; ++nt) {
      union { u32 u[8]; v16bf v; } bf;
      const int nidx = wn * 32 + nt * 16 + ln;
      #pragma unroll
      for (int v = 0; v < 8; ++v)               // VGPR v: K = g*16 + 2v, 2v+1
        bf.u[v] = Bs[cur][(g * 8 + v) * 128 + nidx];
      bfr[nt] = bf.v;
    }

    #pragma unroll
    for (int mt = 0; mt < 4; ++mt)
      #pragma unroll
      for (int nt = 0; nt < 2; ++nt)
        acc[mt][nt] = __builtin_amdgcn_wmma_f32_16x16x32_bf16(
            false, afr[mt], false, bfr[nt], (short)0, acc[mt][nt], false, false);

    if (more) {
      // Buffer nxt was last read one iteration ago (behind the previous barrier),
      // so the async writes above are safe; finish them and publish.
      asm volatile("s_wait_asynccnt 0x0" ::: "memory");
      __syncthreads();
    }
  }

  // ---- epilogue: bias (+relu), write bf16 or f32 ----
  #pragma unroll
  for (int nt = 0; nt < 2; ++nt) {
    const int   n  = bn + wn * 32 + nt * 16 + ln;
    const float bv = bias[n];
    #pragma unroll
    for (int mt = 0; mt < 4; ++mt) {
      #pragma unroll
      for (int v = 0; v < 8; ++v) {
        const int m = bm + wm * 64 + mt * 16 + g * 8 + v;   // C/D layout: VGPR v -> M = g*8+v
        float x = acc[mt][nt][v] + bv;
        if (RELU) x = fmaxf(x, 0.0f);
        if (OUT_BF16) ((u16*)OutP)[(size_t)m * N + n] = f2bf(x);
        else          ((float*)OutP)[(size_t)m * N + n] = x;
      }
    }
  }
}

// ---------------- launch ----------------
extern "C" void kernel_launch(void* const* d_in, const int* in_sizes, int n_in,
                              void* d_out, int out_size, void* d_ws, size_t ws_size,
                              hipStream_t stream) {
  const int   BS = 32 * 1024;        // 32768 token rows
  const int   E  = 512, C = 512, H = 1024, L = 128;

  const int*   tok    = (const int*)  d_in[0];
  const float* emb    = (const float*)d_in[1];
  const float* conv_w = (const float*)d_in[2];
  const float* conv_b = (const float*)d_in[3];
  const float* w1     = (const float*)d_in[4];
  const float* b1     = (const float*)d_in[5];
  const float* w2     = (const float*)d_in[6];
  const float* b2     = (const float*)d_in[7];
  float*       out    = (float*)d_out;

  char* ws = (char*)d_ws;
  u16* X   = (u16*)ws;  ws += (size_t)BS * E * 2;       //  32 MB
  u16* Hb  = (u16*)ws;  ws += (size_t)BS * C * 2;       //  32 MB
  u16* Zb  = (u16*)ws;  ws += (size_t)BS * H * 2;       //  64 MB
  u32* W1p = (u32*)ws;  ws += (size_t)(E / 2) * C * 4;  // 512 KB  (conv_w^T packed)
  u32* W2p = (u32*)ws;  ws += (size_t)(C / 2) * H * 4;  //   1 MB
  u32* W3p = (u32*)ws;  ws += (size_t)(H / 2) * L * 4;  // 256 KB

  // 1) embedding gather + bf16 convert
  gather_embed_bf16<<<(BS * E / 8) / 256, 256, 0, stream>>>(tok, emb, X);

  // 2) weight converts into pre-packed WMMA-B layout
  pack_tcvt_bf16<<<((E / 2) * C + 255) / 256, 256, 0, stream>>>(conv_w, W1p, E / 2, C, E);
  pack_cvt_bf16 <<<((C / 2) * H + 255) / 256, 256, 0, stream>>>(w1, W2p, C / 2, H);
  pack_cvt_bf16 <<<((H / 2) * L + 255) / 256, 256, 0, stream>>>(w2, W3p, H / 2, L);

  // 3) three fused GEMMs
  gemm_bf16_wmma<true,  true ><<<dim3(BS / 128, C / 128), 256, 0, stream>>>(X,  W1p, conv_b, Hb,  BS, C, E);
  gemm_bf16_wmma<true,  true ><<<dim3(BS / 128, H / 128), 256, 0, stream>>>(Hb, W2p, b1,     Zb,  BS, H, C);
  gemm_bf16_wmma<false, false><<<dim3(BS / 128, L / 128), 256, 0, stream>>>(Zb, W3p, b2,     out, BS, L, H);
}
